// GAL_10831907520713
// MI455X (gfx1250) — matrix operand order repaired and checked by reading, not compiled, not run on hardware
//
#include <hip/hip_runtime.h>
#include <hip/hip_bf16.h>

#define N_NODES 100000
#define DIM 128
#define N_EDGES 1600000
#define BATCH 8192
#define BN_EPS 1e-5f
#define APAD 4   // LDS row padding (floats) -> row stride 132 = conflict-free, 16B-aligned

typedef __attribute__((ext_vector_type(16))) _Float16 v16h;
typedef __attribute__((ext_vector_type(8)))  _Float16 v8h;
typedef __attribute__((ext_vector_type(8)))  float    v8f;

// ---------------- utility ----------------
__global__ void k_zero(float* __restrict__ p, int n) {
    int i = blockIdx.x * blockDim.x + threadIdx.x;
    if (i < n) p[i] = 0.0f;
}

// ---------------- BatchNorm stats (fold into per-column scale/shift) ----------------
__global__ void k_bn_stats(const float* __restrict__ X, float* __restrict__ s1,
                           float* __restrict__ s2) {
    // blockDim = 128 (one thread per column), grid-stride over rows; coalesced.
    const int c = threadIdx.x;
    float a = 0.0f, b = 0.0f;
    for (int r = blockIdx.x; r < N_NODES; r += gridDim.x) {
        float v = X[(size_t)r * DIM + c];
        a += v;
        b += v * v;
    }
    atomicAdd(&s1[c], a);
    atomicAdd(&s2[c], b);
}

__global__ void k_bn_coeff(const float* __restrict__ s1, const float* __restrict__ s2,
                           const float* __restrict__ gamma, const float* __restrict__ beta,
                           float* __restrict__ scale, float* __restrict__ shift) {
    const int c = threadIdx.x;  // 128 threads
    const float inv_n = 1.0f / (float)N_NODES;
    float mu  = s1[c] * inv_n;
    float var = s2[c] * inv_n - mu * mu;
    float sc  = gamma[c] * rsqrtf(var + BN_EPS);
    scale[c] = sc;
    shift[c] = beta[c] - mu * sc;
}

// ---------------- transpose+convert weights to f16 (B operand: Wt[n][k]) ----------------
__global__ void k_wt(const float* __restrict__ W, _Float16* __restrict__ Wt) {
    int i = blockIdx.x * blockDim.x + threadIdx.x;  // 128*128
    if (i >= DIM * DIM) return;
    int n = i >> 7, k = i & (DIM - 1);
    Wt[n * DIM + k] = (_Float16)W[k * DIM + n];
}

// ---------------- WMMA GEMM: Y = [BN](X) @ W   (X: Nx128 f32, Wt: f16 transposed) ----------------
// Block = 8 waves; one 16-row M stripe per block; A tile staged to LDS once via
// GLOBAL_LOAD_ASYNC_TO_LDS_B128 (ASYNCcnt), then 4 x v_wmma_f32_16x16x32_f16 per wave.
template <bool BN>
__global__ __launch_bounds__(256) void k_gemm(const float* __restrict__ X,
                                              const _Float16* __restrict__ Wt,
                                              const float* __restrict__ scale,
                                              const float* __restrict__ shift,
                                              float* __restrict__ Y) {
    __shared__ float tileA[16 * (DIM + APAD)];   // 8448 B, padded rows

    // --- cooperative async fill: 512 x 16B chunks, 2 per thread ---
    {
        #pragma unroll
        for (int c = 0; c < 2; c++) {
            unsigned chunk = threadIdx.x + c * 256;          // 0..511
            unsigned row = chunk >> 5;                       // 16 rows, 32 chunks/row
            unsigned col = (chunk & 31u) * 4u;               // float col, 16B chunks
            const float* gp = X + (size_t)(blockIdx.x * 16 + row) * DIM + col;
            unsigned lo = (unsigned)(uintptr_t)&tileA[row * (DIM + APAD) + col];
            asm volatile("global_load_async_to_lds_b128 %0, %1, off"
                         :: "v"(lo), "v"(gp) : "memory");
        }
        asm volatile("s_wait_asynccnt 0" ::: "memory");
    }
    __syncthreads();

    const int lane  = threadIdx.x & 31;
    const int wave  = threadIdx.x >> 5;   // n-tile 0..7 (covers all 128 cols)
    const int mTile = blockIdx.x;         // 0..6249 (100000/16 exact)
    const int mr    = lane & 15;
    const int hi    = lane >> 4;          // half-wave select

    const float* xp = &tileA[mr * (DIM + APAD)];              // A row (LDS)
    const _Float16* wp = Wt + (size_t)(wave * 16 + mr) * DIM; // B "row" (column n of W)

    v8f acc = {};
    #pragma unroll
    for (int kc = 0; kc < DIM; kc += 32) {
        // --- A tile, 16-bit A layout: lane<16 -> K kc+0..7 / kc+16..23 ; lane>=16 -> +8
        const int k0 = kc + hi * 8;
        v16h a;
        #pragma unroll
        for (int j = 0; j < 8; j++) {
            float x0 = xp[k0 + j];
            float x1 = xp[k0 + 16 + j];
            if (BN) {
                x0 = x0 * scale[k0 + j]      + shift[k0 + j];
                x1 = x1 * scale[k0 + 16 + j] + shift[k0 + 16 + j];
            }
            a[j]     = (_Float16)x0;
            a[j + 8] = (_Float16)x1;
        }
        // --- B tile: same K order as A: lane<16 -> K kc+0..7 / kc+16..23 ; lane>=16 -> +8
        v8h b0 = *(const v8h*)(wp + kc + hi * 8);
        v8h b1 = *(const v8h*)(wp + kc + 16 + hi * 8);
        v16h b = __builtin_shufflevector(b0, b1, 0, 1, 2, 3, 4, 5, 6, 7, 8, 9, 10, 11, 12, 13, 14, 15);

        acc = __builtin_amdgcn_wmma_f32_16x16x32_f16(false, a, false, b,
                                                     (short)0, acc, false, false);
    }
    // D layout: col = wave*16 + (lane&15); row = mTile*16 + (lane>=16?8:0) + r
    const int col   = wave * 16 + mr;
    const int mBase = mTile * 16 + hi * 8;
    #pragma unroll
    for (int r = 0; r < 8; r++)
        Y[(size_t)(mBase + r) * DIM + col] = acc[r];
}

// ---------------- degree / dinv ----------------
__global__ void k_deg_init(float* __restrict__ deg) {
    int i = blockIdx.x * blockDim.x + threadIdx.x;
    if (i < N_NODES) deg[i] = 1.0f;   // self loop
}
__global__ void k_deg_edges(const int* __restrict__ dst, float* __restrict__ deg) {
    for (int e = blockIdx.x * blockDim.x + threadIdx.x; e < N_EDGES; e += gridDim.x * blockDim.x)
        atomicAdd(&deg[dst[e]], 1.0f);
}
__global__ void k_dinv(float* __restrict__ deg) {
    int i = blockIdx.x * blockDim.x + threadIdx.x;
    if (i < N_NODES) deg[i] = rsqrtf(deg[i]);   // deg >= 1 always (self loop)
}

// ---------------- propagation: H[d] = sum_e XW[s]*dinv[s]*dinv[d] + XW[d]*dinv[d]^2 + b ----------------
__global__ void k_prop_init(const float* __restrict__ XW, const float* __restrict__ dinv,
                            const float* __restrict__ bias, float* __restrict__ H) {
    size_t i = (size_t)blockIdx.x * blockDim.x + threadIdx.x;  // 50000 blocks * 256 exact
    int node = (int)(i >> 7), c = (int)(i & (DIM - 1));
    float w = dinv[node];
    H[i] = XW[i] * w * w + bias[c];
}

__global__ __launch_bounds__(256) void k_prop_edges(const float* __restrict__ XW,
                                                    const int* __restrict__ src,
                                                    const int* __restrict__ dst,
                                                    const float* __restrict__ dinv,
                                                    float* __restrict__ H) {
    const int c   = threadIdx.x & (DIM - 1);
    const int sub = threadIdx.x >> 7;        // 2 edges per 256-thread block
    const int stride = gridDim.x * 2;
    for (int e = blockIdx.x * 2 + sub; e < N_EDGES; e += stride) {
        int s = src[e], d = dst[e];
        int eNext = e + stride;
        if (eNext < N_EDGES)  // emits global_prefetch_b8 for the next source row
            __builtin_prefetch(&XW[(size_t)src[eNext] * DIM + c], 0, 1);
        float w = dinv[s] * dinv[d];
        atomicAdd(&H[(size_t)d * DIM + c], XW[(size_t)s * DIM + c] * w);
    }
}

// ---------------- MLP heads + gender-mask accumulation ----------------
template <int C>
__device__ __forceinline__ void mlp_head(const float* __restrict__ u,
                                         const float* __restrict__ W1, const float* __restrict__ b1,
                                         const float* __restrict__ W2, const float* __restrict__ b2,
                                         const float* __restrict__ W3, const float* __restrict__ b3,
                                         float* __restrict__ outrow) {
    float h1[32], h2[32];
    for (int j = 0; j < 32; j++) {
        float acc = b1[j];
        #pragma unroll 8
        for (int k = 0; k < DIM; k++) acc += u[k] * W1[k * 32 + j];
        h1[j] = acc > 0.0f ? acc : 0.01f * acc;
    }
    for (int j = 0; j < 32; j++) {
        float acc = b2[j];
        #pragma unroll 8
        for (int k = 0; k < 32; k++) acc += h1[k] * W2[k * 32 + j];
        h2[j] = acc > 0.0f ? acc : 0.01f * acc;
    }
    for (int j = 0; j < C; j++) {
        float acc = b3[j];
        #pragma unroll 8
        for (int k = 0; k < 32; k++) acc += h2[k] * W3[k * C + j];
        outrow[j] = acc;
    }
}

__global__ __launch_bounds__(256) void k_heads(
    const float* __restrict__ H, const int* __restrict__ users, const int* __restrict__ gender,
    const float* __restrict__ Wg1, const float* __restrict__ bg1,
    const float* __restrict__ Wg2, const float* __restrict__ bg2,
    const float* __restrict__ Wg3, const float* __restrict__ bg3,
    const float* __restrict__ Wa1, const float* __restrict__ ba1,
    const float* __restrict__ Wa2, const float* __restrict__ ba2,
    const float* __restrict__ Wa3, const float* __restrict__ ba3,
    const float* __restrict__ Wo1, const float* __restrict__ bo1,
    const float* __restrict__ Wo2, const float* __restrict__ bo2,
    const float* __restrict__ Wo3, const float* __restrict__ bo3,
    float* __restrict__ out, float* __restrict__ avg_ws) {
    int i = blockIdx.x * blockDim.x + threadIdx.x;
    if (i >= BATCH) return;
    const float* u = H + (size_t)users[i] * DIM;

    float* age_out = out + 1;
    float* gen_out = out + 1 + BATCH * 7;
    float* occ_out = out + 1 + BATCH * 7 + BATCH * 2;

    float g[2];
    mlp_head<2>(u, Wg1, bg1, Wg2, bg2, Wg3, bg3, g);
    gen_out[i * 2 + 0] = g[0];
    gen_out[i * 2 + 1] = g[1];
    int gi = gender[i];
    atomicAdd(&avg_ws[gi], g[gi]);   // one-hot masked sum

    float a[7];
    mlp_head<7>(u, Wa1, ba1, Wa2, ba2, Wa3, ba3, a);
    #pragma unroll
    for (int j = 0; j < 7; j++) age_out[i * 7 + j] = a[j];

    float o[21];
    mlp_head<21>(u, Wo1, bo1, Wo2, bo2, Wo3, bo3, o);
    #pragma unroll
    for (int j = 0; j < 21; j++) occ_out[i * 21 + j] = o[j];
}

__global__ void k_loss(const float* __restrict__ avg_ws, float* __restrict__ out) {
    float a0 = avg_ws[0] / (float)BATCH;
    float a1 = avg_ws[1] / (float)BATCH;
    out[0] = fabsf(fmaxf(a0, a1) - fminf(a0, a1));
}

// ---------------- driver ----------------
extern "C" void kernel_launch(void* const* d_in, const int* in_sizes, int n_in,
                              void* d_out, int out_size, void* d_ws, size_t ws_size,
                              hipStream_t stream) {
    const float* embed = (const float*)d_in[0];
    const float* gamma = (const float*)d_in[1];
    const float* beta  = (const float*)d_in[2];
    const float* W1 = (const float*)d_in[3];
    const float* b1 = (const float*)d_in[4];
    const float* W2 = (const float*)d_in[5];
    const float* b2 = (const float*)d_in[6];
    const float* Wg1 = (const float*)d_in[7];  const float* bg1 = (const float*)d_in[8];
    const float* Wg2 = (const float*)d_in[9];  const float* bg2 = (const float*)d_in[10];
    const float* Wg3 = (const float*)d_in[11]; const float* bg3 = (const float*)d_in[12];
    const float* Wa1 = (const float*)d_in[13]; const float* ba1 = (const float*)d_in[14];
    const float* Wa2 = (const float*)d_in[15]; const float* ba2 = (const float*)d_in[16];
    const float* Wa3 = (const float*)d_in[17]; const float* ba3 = (const float*)d_in[18];
    const float* Wo1 = (const float*)d_in[19]; const float* bo1 = (const float*)d_in[20];
    const float* Wo2 = (const float*)d_in[21]; const float* bo2 = (const float*)d_in[22];
    const float* Wo3 = (const float*)d_in[23]; const float* bo3 = (const float*)d_in[24];
    const int* edge_index = (const int*)d_in[25];
    const int* users  = (const int*)d_in[26];
    const int* gender = (const int*)d_in[27];
    const int* src = edge_index;             // row 0
    const int* dst = edge_index + N_EDGES;   // row 1
    float* out = (float*)d_out;

    // ---- workspace layout (base assumed >=16B aligned) ----
    float* bufA = (float*)d_ws;                               // 100000*128 f32
    float* bufB = bufA + (size_t)N_NODES * DIM;               // 100000*128 f32
    float* deg  = bufB + (size_t)N_NODES * DIM;               // 100000 f32 (becomes dinv)
    _Float16* Wt1 = (_Float16*)(deg + N_NODES);               // 128*128 f16 (32B-aligned)
    _Float16* Wt2 = Wt1 + DIM * DIM;
    float* s1    = (float*)(Wt2 + DIM * DIM);                 // 128
    float* s2    = s1 + DIM;                                  // 128
    float* scale = s2 + DIM;                                  // 128
    float* shift = scale + DIM;                               // 128
    float* avg   = shift + DIM;                               // 2

    // zero stats + avg accumulators (s1..avg contiguous: 514 floats)
    k_zero<<<3, 256, 0, stream>>>(s1, 4 * DIM + 2);

    // BatchNorm folded coefficients
    k_bn_stats<<<512, 128, 0, stream>>>(embed, s1, s2);
    k_bn_coeff<<<1, 128, 0, stream>>>(s1, s2, gamma, beta, scale, shift);

    // f16 transposed weights for the WMMA B operand
    k_wt<<<(DIM * DIM + 255) / 256, 256, 0, stream>>>(W1, Wt1);
    k_wt<<<(DIM * DIM + 255) / 256, 256, 0, stream>>>(W2, Wt2);

    // degrees -> dinv
    k_deg_init<<<(N_NODES + 255) / 256, 256, 0, stream>>>(deg);
    k_deg_edges<<<1024, 256, 0, stream>>>(dst, deg);
    k_dinv<<<(N_NODES + 255) / 256, 256, 0, stream>>>(deg);

    const int mTiles = N_NODES / 16;                 // 6250, exact
    const int propInitBlocks = (N_NODES * DIM) / 256; // 50000, exact

    // conv1: XW1 = BN(X) @ W1  (fused BN), then propagate (+b1)
    k_gemm<true><<<mTiles, 256, 0, stream>>>(embed, Wt1, scale, shift, bufA);
    k_prop_init<<<propInitBlocks, 256, 0, stream>>>(bufA, deg, b1, bufB);
    k_prop_edges<<<8192, 256, 0, stream>>>(bufA, src, dst, deg, bufB);

    // conv2: XW2 = H1 @ W2, then propagate (+b2)
    k_gemm<false><<<mTiles, 256, 0, stream>>>(bufB, Wt2, scale, shift, bufA);
    k_prop_init<<<propInitBlocks, 256, 0, stream>>>(bufA, deg, b2, bufB);
    k_prop_edges<<<8192, 256, 0, stream>>>(bufA, src, dst, deg, bufB);

    // heads + masked gender accumulation
    k_heads<<<BATCH / 256, 256, 0, stream>>>(bufB, users, gender,
                                             Wg1, bg1, Wg2, bg2, Wg3, bg3,
                                             Wa1, ba1, Wa2, ba2, Wa3, ba3,
                                             Wo1, bo1, Wo2, bo2, Wo3, bo3,
                                             out, avg);
    k_loss<<<1, 1, 0, stream>>>(avg, out);
}
